// EMLLoss_17884243820948
// MI455X (gfx1250) — matrix operand order repaired and checked by reading, not compiled, not run on hardware
//
#include <hip/hip_runtime.h>
#include <hip/hip_bf16.h>
#include <math.h>

// ---- types -----------------------------------------------------------------
typedef __bf16 v16bf __attribute__((ext_vector_type(16)));
typedef float  v8f   __attribute__((ext_vector_type(8)));
typedef float  f32x4 __attribute__((ext_vector_type(4)));
typedef union { f32x4 q[2]; v16bf v; } frag_t;

// ---- problem constants -----------------------------------------------------
#define NB      8192      // batch rows
#define DIM     512       // embedding dim
#define NLAB    128       // label bins
#define SPLIT   8         // column splits (parallelism over j)
#define WG_ROWS 128       // rows per workgroup (8 waves x 16)
#define JT      64        // columns per j-tile (4 wmma tiles wide)
#define KC      32        // K per wmma (bf16 16x16x32)
#define NTILES  ((NB / SPLIT) / JT)          // 16 j-tiles per workgroup
#define A_BYTES (WG_ROWS * DIM * 2)          // 128 KB
#define B_BYTES (JT * DIM * 2)               // 64 KB per buffer
#define EPSV    1e-8f

// ---- CDNA5 async LDS DMA helpers -------------------------------------------
// GLOBAL_LOAD_ASYNC_TO_LDS_B128: VDST = 32-bit LDS byte offset (VGPR),
// VADDR = 64-bit global address (VGPR pair). Tracked by ASYNCcnt.
__device__ __forceinline__ void async_b128(unsigned ldsoff, const void* g) {
    unsigned long long ga = (unsigned long long)g;
    asm volatile("global_load_async_to_lds_b128 %0, %1, off"
                 :: "v"(ldsoff), "v"(ga) : "memory");
}
__device__ __forceinline__ void wait_async0() {
    asm volatile("s_wait_asynccnt 0x0" ::: "memory");
}

// ============================================================================
// Kernel 0: zero label histogram
// ============================================================================
__global__ void eml_init(int* __restrict__ hist) {
    if (threadIdx.x < NLAB) hist[threadIdx.x] = 0;
}

// ============================================================================
// Kernel 1: L2-normalize rows -> bf16, build label histogram.
// One wave (32 lanes) per row; 16 floats per lane.
// ============================================================================
__global__ __launch_bounds__(256)
void eml_norm(const float* __restrict__ emb, const int* __restrict__ labels,
              __bf16* __restrict__ ebf, int* __restrict__ hist) {
    const int tid  = threadIdx.x;
    const int wave = tid >> 5;
    const int lane = tid & 31;
    const int row  = blockIdx.x * 8 + wave;

    const float* src = emb + (size_t)row * DIM + lane * 16;
    f32x4 v[4];
    float ss = 0.f;
#pragma unroll
    for (int q = 0; q < 4; ++q) {
        v[q] = *(const f32x4*)(src + q * 4);
#pragma unroll
        for (int e = 0; e < 4; ++e) ss += v[q][e] * v[q][e];
    }
#pragma unroll
    for (int m = 16; m >= 1; m >>= 1) ss += __shfl_xor(ss, m, 32);

    const float nrm = fmaxf(sqrtf(ss), 1e-12f);
    const float s   = 1.0f / nrm;

    __bf16* dst = ebf + (size_t)row * DIM + lane * 16;
    union { __bf16 h[8]; f32x4 q; } p0, p1;
#pragma unroll
    for (int e = 0; e < 4; ++e) {
        p0.h[e]     = (__bf16)(v[0][e] * s);
        p0.h[4 + e] = (__bf16)(v[1][e] * s);
        p1.h[e]     = (__bf16)(v[2][e] * s);
        p1.h[4 + e] = (__bf16)(v[3][e] * s);
    }
    *(f32x4*)(dst)     = p0.q;
    *(f32x4*)(dst + 8) = p1.q;

    if (lane == 0) atomicAdd(&hist[labels[row] & (NLAB - 1)], 1);
}

// ============================================================================
// Kernel 2: fused bf16 WMMA Gram-matrix + masked exp row reductions.
// Grid: (NB/WG_ROWS)*SPLIT workgroups of 256 threads (8 waves).
// LDS (256 KB): A strip 128x512 bf16 staged once via async DMA,
//               B tile 64x512 bf16 double-buffered via async DMA overlapped
//               with WMMA compute. K loop software-pipelined over fragments.
// 256 KB LDS => exactly 1 WG per WGP => 2 waves/SIMD; tell the compiler so
// it can keep both fragment register sets live instead of dscnt-0 reuse.
// ============================================================================
__global__ __launch_bounds__(256)
__attribute__((amdgpu_waves_per_eu(2)))
void eml_gemm(const __bf16* __restrict__ ebf, const int* __restrict__ labels,
              float* __restrict__ posPart, float* __restrict__ negPart) {
    extern __shared__ char smem[];
    __bf16* sA = (__bf16*)smem;                       // 128 x 512
    __bf16* sB = (__bf16*)(smem + A_BYTES);           // 2 x (64 x 512)

    const unsigned sAo = (unsigned)(uintptr_t)sA;     // LDS byte offsets
    const unsigned sBo = (unsigned)(uintptr_t)sB;

    const int strip = blockIdx.x / SPLIT;             // 0..63
    const int split = blockIdx.x % SPLIT;             // 0..7
    const int i0    = strip * WG_ROWS;
    const int jBase = split * (NB / SPLIT);           // 1024-column slab
    const int tid   = threadIdx.x;
    const int wave  = tid >> 5;
    const int lane  = tid & 31;
    const int hsel  = lane >> 4;
    const int l15   = lane & 15;

    // ---- async-stage full A strip + first B tile into LDS ----
    {
        const __bf16* gA = ebf + (size_t)i0 * DIM;
        for (int c = tid; c < A_BYTES / 16; c += 256)
            async_b128(sAo + (unsigned)c * 16u, gA + c * 8);
        const __bf16* gB = ebf + (size_t)jBase * DIM;
        for (int c = tid; c < B_BYTES / 16; c += 256)
            async_b128(sBo + (unsigned)c * 16u, gB + c * 8);
    }

    // C-layout: VGPR r of lane L -> row M = r + 8*(L>>4), col N = L&15
    const int rowBase = i0 + wave * 16 + hsel * 8;
    int rlab[8];
#pragma unroll
    for (int r = 0; r < 8; ++r) rlab[r] = labels[rowBase + r];

    float accP[8], accN[8];
#pragma unroll
    for (int r = 0; r < 8; ++r) { accP[r] = 0.f; accN[r] = 0.f; }

    wait_async0();
    __syncthreads();

    const __bf16* aBase = sA + (wave * 16 + l15) * DIM + hsel * 8;

    for (int jt = 0; jt < NTILES; ++jt) {
        const int j0  = jBase + jt * JT;
        const int buf = jt & 1;

        // ---- async DMA of the NEXT B tile, overlapped with compute ----
        if (jt + 1 < NTILES) {
            const unsigned dsto = sBo + (unsigned)(buf ^ 1) * B_BYTES;
            const __bf16*  gB   = ebf + (size_t)(j0 + JT) * DIM;
            for (int c = tid; c < B_BYTES / 16; c += 256)
                async_b128(dsto + (unsigned)c * 16u, gB + c * 8);
        }
        if (jt + 2 < NTILES)   // warm L2 two tiles ahead (global_prefetch_b8)
            __builtin_prefetch(ebf + (size_t)(j0 + 2 * JT) * DIM + tid * 64, 0, 1);

        const __bf16* sBc   = sB + buf * (JT * DIM);
        const __bf16* bBase = sBc + l15 * DIM + hsel * 8;

        v8f c0 = {0.f,0.f,0.f,0.f,0.f,0.f,0.f,0.f};
        v8f c1 = c0, c2 = c0, c3 = c0;

        // ---- software-pipelined K loop: load chunk ki+1 while ki computes ----
        frag_t fa[2], fb[2][4];
        fa[0].q[0] = *(const f32x4*)(aBase);
        fa[0].q[1] = *(const f32x4*)(aBase + 16);
#pragma unroll
        for (int t = 0; t < 4; ++t) {
            const __bf16* p = bBase + t * 16 * DIM;
            fb[0][t].q[0] = *(const f32x4*)p;
            fb[0][t].q[1] = *(const f32x4*)(p + 16);
        }
#pragma unroll
        for (int ki = 0; ki < DIM / KC; ++ki) {
            const int cur = ki & 1, nxt = cur ^ 1;
            if (ki + 1 < DIM / KC) {
                const int kn = (ki + 1) * KC;
                fa[nxt].q[0] = *(const f32x4*)(aBase + kn);
                fa[nxt].q[1] = *(const f32x4*)(aBase + kn + 16);
#pragma unroll
                for (int t = 0; t < 4; ++t) {
                    const __bf16* p = bBase + t * 16 * DIM + kn;
                    fb[nxt][t].q[0] = *(const f32x4*)p;
                    fb[nxt][t].q[1] = *(const f32x4*)(p + 16);
                }
            }
            c0 = __builtin_amdgcn_wmma_f32_16x16x32_bf16(
                     false, fa[cur].v, false, fb[cur][0].v, (short)0, c0, false, false);
            c1 = __builtin_amdgcn_wmma_f32_16x16x32_bf16(
                     false, fa[cur].v, false, fb[cur][1].v, (short)0, c1, false, false);
            c2 = __builtin_amdgcn_wmma_f32_16x16x32_bf16(
                     false, fa[cur].v, false, fb[cur][2].v, (short)0, c2, false, false);
            c3 = __builtin_amdgcn_wmma_f32_16x16x32_bf16(
                     false, fa[cur].v, false, fb[cur][3].v, (short)0, c3, false, false);
        }

        // ---- epilogue: exp(g-1), label masks, accumulate per-row ----
#pragma unroll
        for (int t = 0; t < 4; ++t) {
            const v8f ct = (t == 0) ? c0 : (t == 1) ? c1 : (t == 2) ? c2 : c3;
            const int j  = j0 + t * 16 + l15;
            const int jl = labels[j];
#pragma unroll
            for (int r = 0; r < 8; ++r) {
                const float x = __expf(ct[r] - 1.0f);
                if (jl == rlab[r]) {
                    if (j != rowBase + r) accP[r] += x;   // positive, no diagonal
                } else {
                    accN[r] += x;                         // negative
                }
            }
        }

        // our async loads for the next buffer are done, then everyone syncs
        wait_async0();
        __syncthreads();
    }

    // reduce over the 16 lanes that share each row (columns of the tile)
#pragma unroll
    for (int r = 0; r < 8; ++r) {
#pragma unroll
        for (int m = 8; m >= 1; m >>= 1) {
            accP[r] += __shfl_xor(accP[r], m, 16);
            accN[r] += __shfl_xor(accN[r], m, 16);
        }
    }
    if (l15 == 0) {
#pragma unroll
        for (int r = 0; r < 8; ++r) {
            posPart[split * NB + rowBase + r] = accP[r];
            negPart[split * NB + rowBase + r] = accN[r];
        }
    }
}

// ============================================================================
// Kernel 3: combine split partials, per-row loss, reduce to scalar.
// ============================================================================
__global__ __launch_bounds__(256)
void eml_final(const float* __restrict__ posPart, const float* __restrict__ negPart,
               const int* __restrict__ labels, const int* __restrict__ hist,
               float* __restrict__ out) {
    __shared__ float red[256];
    float acc = 0.f;
    for (int i = threadIdx.x; i < NB; i += 256) {
        float p = 0.f, n = 0.f;
#pragma unroll
        for (int s = 0; s < SPLIT; ++s) {
            p += posPart[s * NB + i];
            n += negPart[s * NB + i];
        }
        const int   cnt = hist[labels[i] & (NLAB - 1)];
        const float pc  = (float)(cnt - 1);
        const float nc  = (float)(NB - cnt);
        const float pm  = p / fmaxf(pc, 1.f);
        const float nm  = n / fmaxf(nc, 1.f);
        const bool  ok  = (pc > 0.f) && (nc > 0.f);
        const float l   = -__logf(pm / (pm + nm + EPSV));
        acc += ok ? l : 0.f;
    }
    red[threadIdx.x] = acc;
    __syncthreads();
    for (int s = 128; s >= 1; s >>= 1) {
        if (threadIdx.x < s) red[threadIdx.x] += red[threadIdx.x + s];
        __syncthreads();
    }
    if (threadIdx.x == 0) out[0] = red[0] / (float)NB;
}

// ============================================================================
extern "C" void kernel_launch(void* const* d_in, const int* in_sizes, int n_in,
                              void* d_out, int out_size, void* d_ws, size_t ws_size,
                              hipStream_t stream) {
    (void)in_sizes; (void)n_in; (void)out_size; (void)ws_size;
    const float* emb    = (const float*)d_in[0];
    const int*   labels = (const int*)d_in[1];

    char*   ws      = (char*)d_ws;
    int*    hist    = (int*)ws;                                   // 512 B
    float*  posPart = (float*)(ws + 512);                         // SPLIT*NB f32
    float*  negPart = (float*)(ws + 512 + (size_t)SPLIT * NB * 4);
    __bf16* ebf     = (__bf16*)(ws + 512 + (size_t)2 * SPLIT * NB * 4); // NB*DIM bf16

    eml_init<<<1, NLAB, 0, stream>>>(hist);
    eml_norm<<<NB / 8, 256, 0, stream>>>(emb, labels, ebf, hist);

    const size_t lds = A_BYTES + 2 * B_BYTES;                     // 256 KB
    eml_gemm<<<(NB / WG_ROWS) * SPLIT, 256, lds, stream>>>(ebf, labels, posPart, negPart);

    eml_final<<<1, 256, 0, stream>>>(posPart, negPart, labels, hist, (float*)d_out);
}
